// Rits_i_20615843021474
// MI455X (gfx1250) — compile-verified, matrix operand order checked by reading, not run on hardware
//
#include <hip/hip_runtime.h>
#include <hip/hip_bf16.h>

// RITS-I LSTM imputation for MI455X (gfx1250, wave32, WMMA).
// ~2.2 TFLOP of h[16384,128] @ W_hh^T[128,512] over T=1024 sequential steps,
// mapped onto v_wmma_f32_16x16x32_f16 with the weight (B) operand VGPR-resident.
// Gate-column ownership is interleaved so each wave's accumulators hold matching
// i/f/g/o for its hidden-unit range -> in-register LSTM cell update, no gate
// round trip through LDS. Input chunks staged with global_load_async_to_lds.

#define B_TOT 16384
#define T_LEN 1024
#define H_DIM 128
#define G4    512      // 4*H gate columns
#define MB    16       // batch rows per workgroup
#define CT    32       // time-chunk staged into LDS
#define NT    256      // threads per block (8 wave32s)

#define ASYNC_STAGE 1  // use GLOBAL_LOAD_ASYNC_TO_LDS_B32 for input staging

typedef __attribute__((ext_vector_type(16))) _Float16 v16h;
typedef __attribute__((ext_vector_type(8)))  float    v8f;

__device__ __forceinline__ float fast_tanh(float x) {
#if __has_builtin(__builtin_amdgcn_tanhf)
    return __builtin_amdgcn_tanhf(x);            // v_tanh_f32
#elif __has_builtin(__builtin_amdgcn_tanh_f32)
    return __builtin_amdgcn_tanh_f32(x);
#else
    return 1.f - 2.f / (__expf(2.f * x) + 1.f);
#endif
}
__device__ __forceinline__ float fast_sig(float x) {
    return 0.5f + 0.5f * fast_tanh(0.5f * x);
}

#if ASYNC_STAGE
// Async copy 4B global -> LDS (GV mode), tracked by ASYNCcnt.
__device__ __forceinline__ void async_ld32(void* lds, const void* gaddr) {
    asm volatile("global_load_async_to_lds_b32 %0, %1, off"
                 :: "v"((uint32_t)(uintptr_t)lds),
                    "v"((uint64_t)(uintptr_t)gaddr)
                 : "memory");
}
__device__ __forceinline__ void async_wait0() {
    asm volatile("s_wait_asynccnt 0x0" ::: "memory");
}
#endif

__global__ void rits_init(float* ws) {
    ws[0] = 0.f; ws[1] = 0.f; ws[2] = 0.f;
}

// Column sums of masks: summ[t] = sum_b masks[b,t]  (recurrence-independent)
__global__ void rits_summ(const float* __restrict__ masks, float* __restrict__ summ) {
    __shared__ float red[256];
    const int t = blockIdx.x;
    float s = 0.f;
    for (int b = threadIdx.x; b < B_TOT; b += 256)
        s += masks[(size_t)b * T_LEN + t];
    red[threadIdx.x] = s;
    __syncthreads();
    for (int off = 128; off > 0; off >>= 1) {
        if (threadIdx.x < off) red[threadIdx.x] += red[threadIdx.x + off];
        __syncthreads();
    }
    if (threadIdx.x == 0) summ[t] = red[0];
}

__global__ void rits_fin(const float* __restrict__ ws, float* __restrict__ out) {
    out[0] = ws[0] * 0.3f + (ws[1] / (ws[2] + 1e-5f)) * 1.0f;
}

__global__ __launch_bounds__(NT, 1)
void rits_main(const float* __restrict__ values,
               const float* __restrict__ masks,
               const float* __restrict__ deltas,
               const float* __restrict__ labels,
               const float* __restrict__ is_train,
               const float* __restrict__ Wd,   const float* __restrict__ bd,
               const float* __restrict__ Wih,  const float* __restrict__ bih,
               const float* __restrict__ Whh,  const float* __restrict__ bhh,
               const float* __restrict__ Wreg, const float* __restrict__ breg,
               const float* __restrict__ Wout, const float* __restrict__ bout,
               const float* __restrict__ summ,
               float* __restrict__ out_pred,
               float* __restrict__ out_imp,
               float* __restrict__ ws_acc)
{
    __shared__ float    h32[MB][H_DIM];
    __shared__ _Float16 h16[MB][H_DIM];
    __shared__ float    xh_part[MB][16];
    __shared__ float    xc_l[MB], ml_l[MB], numarr[MB];
    __shared__ float    s_wd[H_DIM], s_bd[H_DIM], s_wreg[H_DIM], s_wout[H_DIM];
    __shared__ float    s_w0[G4], s_w1[G4], s_bsum[G4];
    __shared__ float    vch[MB][CT], mch[MB][CT], dch[MB][CT];
    __shared__ float    sch[CT];

    const int tid  = threadIdx.x;
    const int wave = tid >> 5;
    const int lane = tid & 31;
    const int n    = lane & 15;   // column within 16x16 tile / A row
    const int grp  = lane >> 4;   // lane half-group (K offset selector)
    const int wj0  = wave * 16;   // hidden-unit range base owned by this wave
    const int bg0  = blockIdx.x * MB;

    // ---- one-time parameter staging ----
    if (tid < H_DIM) {
        s_wd[tid]   = Wd[tid];
        s_bd[tid]   = bd[tid];
        s_wreg[tid] = Wreg[tid];
        s_wout[tid] = Wout[tid];
    }
    for (int g = tid; g < G4; g += NT) {
        s_w0[g]   = Wih[2 * g];
        s_w1[g]   = Wih[2 * g + 1];
        s_bsum[g] = bih[g] + bhh[g];
    }
    for (int i = tid; i < MB * H_DIM; i += NT) {
        ((float*)h32)[i]    = 0.f;
        ((_Float16*)h16)[i] = (_Float16)0.f;
    }

    // ---- preload W_hh^T as f16 WMMA B-operands, VGPR-resident for whole loop ----
    // Wave w's N-tile nt covers gate columns nt*128 + w*16 + [0,16):
    // nt=0 -> i, nt=1 -> f, nt=2 -> g, nt=3 -> o, all for j in [16w,16w+16).
    v16h btl[4][4];
#pragma unroll
    for (int nt = 0; nt < 4; ++nt) {
        const int g = nt * H_DIM + wj0 + n;
        const float* wrow = Whh + (size_t)g * H_DIM;
#pragma unroll
        for (int kc = 0; kc < 4; ++kc) {
#pragma unroll
            for (int v = 0; v < 8; ++v) {
                int k = kc * 32 + ((v < 4) ? 0 : 16) + grp * 8 + (v & 3) * 2;
                btl[nt][kc][2 * v]     = (_Float16)wrow[k];
                btl[nt][kc][2 * v + 1] = (_Float16)wrow[k + 1];
            }
        }
    }

    __syncthreads();

    // per-lane gate-column constants for accumulator init
    float w0r[4], w1r[4], bsr[4];
#pragma unroll
    for (int nt = 0; nt < 4; ++nt) {
        int g = nt * H_DIM + wj0 + n;
        w0r[nt] = s_w0[g]; w1r[nt] = s_w1[g]; bsr[nt] = s_bsum[g];
    }

    const int   bb = tid >> 4;        // batch row served in elementwise phase A
    const int   jc = tid & 15;        // j-chunk index
    const int   jb = jc * 8;          // j base (8 consecutive hidden units)
    const float breg0 = breg[0];
    const float bout0 = bout[0];

    float cst[8];                     // cell state for (row r+8*grp, col wj0+n)
#pragma unroll
    for (int q = 0; q < 8; ++q) cst[q] = 0.f;
    float xl_local = 0.f;

    // ================= time loop =================
    for (int t0 = 0; t0 < T_LEN; t0 += CT) {
        // stage input chunk [MB x CT] (coalesced along t)
        for (int idx = tid; idx < MB * CT; idx += NT) {
            int b = idx >> 5, ct = idx & (CT - 1);
            size_t gix = (size_t)(bg0 + b) * T_LEN + (size_t)(t0 + ct);
#if ASYNC_STAGE
            async_ld32(&vch[b][ct], values + gix);
            async_ld32(&mch[b][ct], masks + gix);
            async_ld32(&dch[b][ct], deltas + gix);
#else
            vch[b][ct] = values[gix];
            mch[b][ct] = masks[gix];
            dch[b][ct] = deltas[gix];
#endif
        }
        if (tid < CT) sch[tid] = summ[t0 + tid];
#if ASYNC_STAGE
        async_wait0();
#endif
        __syncthreads();

        for (int ct = 0; ct < CT; ++ct) {
            const int t = t0 + ct;

            // ---- phase A: temporal decay, h *= gamma, x_h partial ----
            {
                float dv = dch[bb][ct];
                float part = 0.f;
#pragma unroll
                for (int q = 0; q < 8; ++q) {
                    int j = jb + q;
                    float r = fmaxf(dv * s_wd[j] + s_bd[j], 0.f);
                    float gam = __expf(-r);
                    float hv = h32[bb][j] * gam;
                    h32[bb][j] = hv;
                    h16[bb][j] = (_Float16)hv;
                    part += hv * s_wreg[j];
                }
                xh_part[bb][jc] = part;
            }
            __syncthreads();

            // ---- phase B: x_h, imputation, loss numerator ----
            if (tid < MB) {
                int b = tid;
                float xh = breg0;
#pragma unroll
                for (int q = 0; q < 16; ++q) xh += xh_part[b][q];
                float x  = vch[b][ct];
                float m  = mch[b][ct];
                float xc = m * x + (1.f - m) * xh;
                out_imp[(size_t)(bg0 + b) * T_LEN + t] = xc;
                xc_l[b]   = xc;
                ml_l[b]   = m;
                numarr[b] = fabsf(x - xh) * m;
            }
            __syncthreads();

            if (tid == 0) {
                float num = 0.f;
#pragma unroll
                for (int q = 0; q < MB; ++q) num += numarr[q];
                xl_local += num / (sch[ct] + 1e-5f);
            }

            // ---- phase C: gates = cell_in@Wih^T + h@Whh^T + biases (WMMA) ----
            v8f acc[4];
#pragma unroll
            for (int nt = 0; nt < 4; ++nt) {
#pragma unroll
                for (int r = 0; r < 8; ++r) {
                    int mr = r + grp * 8;   // C/D layout: row M = vgpr + 8*grp
                    acc[nt][r] = xc_l[mr] * w0r[nt] + ml_l[mr] * w1r[nt] + bsr[nt];
                }
            }
            const _Float16* hrow = &h16[n][0];   // A row = lane&15
#pragma unroll
            for (int kc = 0; kc < 4; ++kc) {
                v16h a;
#pragma unroll
                for (int v = 0; v < 8; ++v) {
                    int k = kc * 32 + ((v < 4) ? 0 : 16) + grp * 8 + (v & 3) * 2;
                    a[2 * v]     = hrow[k];
                    a[2 * v + 1] = hrow[k + 1];
                }
#pragma unroll
                for (int nt = 0; nt < 4; ++nt)
                    acc[nt] = __builtin_amdgcn_wmma_f32_16x16x32_f16(
                        false, a, false, btl[nt][kc], (short)0, acc[nt], false, false);
            }

            // ---- phase D: in-register LSTM cell update ----
            // acc[0]=i, acc[1]=f, acc[2]=g, acc[3]=o for (row r+8*grp, j=wj0+n)
            {
                const int j = wj0 + n;
#pragma unroll
                for (int r = 0; r < 8; ++r) {
                    float si = fast_sig(acc[0][r]);
                    float sf = fast_sig(acc[1][r]);
                    float tg = fast_tanh(acc[2][r]);
                    float so = fast_sig(acc[3][r]);
                    float c  = sf * cst[r] + si * tg;
                    cst[r]   = c;
                    h32[r + grp * 8][j] = so * fast_tanh(c);
                }
            }
            __syncthreads();
        }
    }

    // ================= epilogue: predictions + BCE =================
    {
        float part = 0.f;
#pragma unroll
        for (int q = 0; q < 8; ++q) part += h32[bb][jb + q] * s_wout[jb + q];
        xh_part[bb][jc] = part;
        __syncthreads();
        if (tid < MB) {
            int bgl = bg0 + tid;
            float y = bout0;
#pragma unroll
            for (int q = 0; q < 16; ++q) y += xh_part[tid][q];
            float pred = 1.f / (1.f + __expf(-y));
            out_pred[bgl] = pred;
            float lab = labels[bgl];
            float it  = is_train[bgl];
            float mv  = fmaxf(-y, 0.f);
            float bce = y - y * lab + mv + logf(__expf(-mv) + __expf(-y - mv));
            atomicAdd(&ws_acc[1], bce * it);
            atomicAdd(&ws_acc[2], it);
        }
        if (tid == 0) atomicAdd(&ws_acc[0], xl_local);
    }
}

extern "C" void kernel_launch(void* const* d_in, const int* in_sizes, int n_in,
                              void* d_out, int out_size, void* d_ws, size_t ws_size,
                              hipStream_t stream) {
    const float* values   = (const float*)d_in[0];
    const float* masks    = (const float*)d_in[1];
    const float* deltas   = (const float*)d_in[2];
    // d_in[3] evals, d_in[4] eval_masks: unused by the reference outputs
    const float* labels   = (const float*)d_in[5];
    const float* is_train = (const float*)d_in[6];
    const float* Wd       = (const float*)d_in[7];
    const float* bd       = (const float*)d_in[8];
    const float* Wih      = (const float*)d_in[9];
    const float* bih      = (const float*)d_in[10];
    const float* Whh      = (const float*)d_in[11];
    const float* bhh      = (const float*)d_in[12];
    const float* Wreg     = (const float*)d_in[13];
    const float* breg     = (const float*)d_in[14];
    const float* Wout     = (const float*)d_in[15];
    const float* bout     = (const float*)d_in[16];

    float* out  = (float*)d_out;              // [loss(1) | pred(B) | imp(B*T)]
    float* ws   = (float*)d_ws;               // [0..2]=accums, [16..16+T)=summ
    float* summ = ws + 16;

    rits_init<<<1, 1, 0, stream>>>(ws);
    rits_summ<<<T_LEN, 256, 0, stream>>>(masks, summ);
    rits_main<<<B_TOT / MB, NT, 0, stream>>>(
        values, masks, deltas, labels, is_train,
        Wd, bd, Wih, bih, Whh, bhh, Wreg, breg, Wout, bout,
        summ, out + 1, out + 1 + B_TOT, ws);
    rits_fin<<<1, 1, 0, stream>>>(ws, out);
}
